// CustomMultiHeadAttentionStoich_62466004353099
// MI455X (gfx1250) — matrix operand order repaired and checked by reading, not compiled, not run on hardware
//
#include <hip/hip_runtime.h>
#include <hip/hip_bf16.h>
#include <math.h>

#define BB 8
#define TT 1024
#define DD 512
#define HH 8
#define HD 64
#define QT 32
#define EPSF 1e-5f

typedef __attribute__((ext_vector_type(16))) _Float16 v16h;
typedef __attribute__((ext_vector_type(8)))  float    v8f;

__device__ __forceinline__ v8f wmma16(v16h a, v16h b, v8f c) {
  // D = A(16x32 f16) x B(32x16 f16) + C(f32)
  return __builtin_amdgcn_wmma_f32_16x16x32_f16(false, a, false, b, (short)0, c, false, false);
}

__device__ __forceinline__ v16h cvt16(float4 x0, float4 x1, float4 x2, float4 x3) {
  v16h a;
  a[0]  = (_Float16)x0.x; a[1]  = (_Float16)x0.y; a[2]  = (_Float16)x0.z; a[3]  = (_Float16)x0.w;
  a[4]  = (_Float16)x1.x; a[5]  = (_Float16)x1.y; a[6]  = (_Float16)x1.z; a[7]  = (_Float16)x1.w;
  a[8]  = (_Float16)x2.x; a[9]  = (_Float16)x2.y; a[10] = (_Float16)x2.z; a[11] = (_Float16)x2.w;
  a[12] = (_Float16)x3.x; a[13] = (_Float16)x3.y; a[14] = (_Float16)x3.z; a[15] = (_Float16)x3.w;
  return a;
}

// A-matrix fragment (16x32) from row-major f32 (row stride lda floats).
// Lane holds row M=lane&15; K runs [kb..kb+16),
// kb = 16*(lane>=16)  ->  4 x float4 (b128) loads per lane.
__device__ __forceinline__ v16h load_a_frag_f32(const float* __restrict__ A, int lda) {
  const int l = threadIdx.x & 31;
  const float* row = A + (size_t)(l & 15) * lda + ((l >> 4) << 4);
  float4 x0 = *(const float4*)(row + 0);
  float4 x1 = *(const float4*)(row + 4);
  float4 x2 = *(const float4*)(row + 8);
  float4 x3 = *(const float4*)(row + 12);
  return cvt16(x0, x1, x2, x3);
}

// B-matrix fragment (32x16) where memory holds the transpose Bt[n][k]
// (row-major N x K, e.g. K-projection rows). Lane: col N=lane&15,
// K run = 16*(lane>=16)+e  ->  contiguous 16 floats = 4 x float4.
__device__ __forceinline__ v16h load_bt_frag_f32(const float* __restrict__ Bt, int ld) {
  const int l = threadIdx.x & 31;
  const float* row = Bt + (size_t)(l & 15) * ld + ((l >> 4) << 4);
  float4 x0 = *(const float4*)(row + 0);
  float4 x1 = *(const float4*)(row + 4);
  float4 x2 = *(const float4*)(row + 8);
  float4 x3 = *(const float4*)(row + 12);
  return cvt16(x0, x1, x2, x3);
}

// Packed-B fragment: pre-swizzled f16, 32 contiguous bytes per lane
// (layout [frag][lane][16]) -> two global_load_b128, no converts.
__device__ __forceinline__ v16h load_b_packed(const _Float16* __restrict__ P, size_t frag) {
  return *(const v16h*)(P + (frag * 32 + (threadIdx.x & 31)) * 16);
}

// -------- Pack a rows x 512 f32 matrix into f16 B-fragment layout -----------
// Fragment grid: kt = row/32, nt = col/16; frag index = kt*(DD/16)+nt.
// Lane holds col n=lane&15, K run kb=16*(lane>=16)+e. One thread = one lane
// of one fragment: 16 strided reads (done once), one 32B vector store.
__global__ void pack_b_kernel(const float* __restrict__ M, _Float16* __restrict__ P,
                              int rows) {
  const int gid = blockIdx.x * blockDim.x + threadIdx.x;
  const int nFrag = (rows / 32) * (DD / 16);
  if (gid >= nFrag * 32) return;
  const int lane = gid & 31;
  const int frag = gid >> 5;
  const int nt = frag % (DD / 16);
  const int kt = frag / (DD / 16);
  const int n  = lane & 15;
  const int kb = (lane >> 4) << 4;
  const float* src = M + (size_t)(kt * 32 + kb) * DD + nt * 16 + n;
  v16h v;
#pragma unroll
  for (int e = 0; e < 16; ++e) v[e] = (_Float16)src[(size_t)e * DD];
  *(v16h*)(P + (size_t)gid * 16) = v;
}

// -------- Projection GEMM: Y[nrows x 512] = X @ W + bias --------------------
// One wave computes a 32x64 slab: 2 A-frags x 4 packed-B frags = 8 WMMA/k-step.
__global__ void proj_kernel(const float* __restrict__ X, const _Float16* __restrict__ Wp,
                            const float* __restrict__ bias, float* __restrict__ Y,
                            int nrows) {
  const int wave = (int)((blockIdx.x * blockDim.x + threadIdx.x) >> 5);
  const int lane = threadIdx.x & 31;
  const int colGroups = DD / 64;                 // 8
  const int rowTile = wave / colGroups;          // 32-row tiles
  const int colGrp  = wave % colGroups;
  const int r0 = rowTile * 32;
  if (r0 >= nrows) return;
  const int c0 = colGrp * 64;

  v8f acc[2][4] = {};
  for (int k0 = 0; k0 < DD; k0 += 32) {
    v16h a0 = load_a_frag_f32(X + (size_t)r0 * DD + k0, DD);
    v16h a1 = load_a_frag_f32(X + (size_t)(r0 + 16) * DD + k0, DD);
    const size_t kt = (size_t)(k0 >> 5);
#pragma unroll
    for (int t = 0; t < 4; ++t) {
      v16h b = load_b_packed(Wp, kt * (DD / 16) + (c0 >> 4) + t);
      acc[0][t] = wmma16(a0, b, acc[0][t]);
      acc[1][t] = wmma16(a1, b, acc[1][t]);
    }
  }
  const int n  = lane & 15;
  const int rb = (lane >> 4) << 3;
#pragma unroll
  for (int g = 0; g < 2; ++g)
#pragma unroll
    for (int t = 0; t < 4; ++t) {
      const float bv = bias[c0 + t * 16 + n];
#pragma unroll
      for (int r = 0; r < 8; ++r)
        Y[(size_t)(r0 + g * 16 + rb + r) * DD + c0 + t * 16 + n] = acc[g][t][r] + bv;
    }
}

// -------- Per-(b,q) stoich statistics: SumP, SumN, SumP^2, SumN^2 -----------
// relu(Dm) and -relu(-Dm) have disjoint support -> per-head mean/var of the
// stoich bias are affine in these 4 scalars; no T x T pass per head needed.
__global__ void stoich_stats_kernel(const float* __restrict__ frac,
                                    float* __restrict__ stats) {
  int idx = blockIdx.x * blockDim.x + threadIdx.x;  // b*T + q
  if (idx >= BB * TT) return;
  int b = idx / TT;
  float fi = frac[idx];
  const float* fr = frac + (size_t)b * TT;
  float sP = 0.f, sN = 0.f, sP2 = 0.f, sN2 = 0.f;
  for (int k = 0; k < TT; ++k) {
    float fj = fr[k];
    float dm = (fj - fi) * fi * fj;
    float p  = dm > 0.f ? dm : 0.f;
    float nn = dm < 0.f ? dm : 0.f;
    sP += p; sN += nn; sP2 += p * p; sN2 += nn * nn;
  }
  stats[idx * 4 + 0] = sP;  stats[idx * 4 + 1] = sN;
  stats[idx * 4 + 2] = sP2; stats[idx * 4 + 3] = sN2;
}

// -------- Fused attention: one block = (b, h, 32-query strip) ---------------
// Score strip S[32][1024] stays resident in LDS (128 KB of the 320 KB WGP LDS);
// V is consumed as pre-packed f16 fragments (2 x b128 per WMMA).
__global__ void attn_kernel(const float* __restrict__ Qp, const float* __restrict__ Kp,
                            const _Float16* __restrict__ Vpk,
                            const float* __restrict__ frac,
                            const float* __restrict__ stats,
                            const float* __restrict__ alpha_pos,
                            const float* __restrict__ alpha_neg,
                            const float* __restrict__ gamma_p,
                            const float* __restrict__ delta_p,
                            const int* __restrict__ addb_p,
                            float* __restrict__ Ob) {
  extern __shared__ float smem[];
  float* S    = smem;                 // QT*TT
  float* redA = S    + QT * TT;       // QT*8
  float* redB = redA + QT * 8;        // QT*8
  float* muL  = redB + QT * 8;        // QT
  float* isL  = muL  + QT;
  float* muS  = isL  + QT;
  float* isS  = muS  + QT;
  float* rmax = isS  + QT;
  float* rden = rmax + QT;

  const int blk = blockIdx.x;
  const int qt  = blk % (TT / QT);
  const int tmp = blk / (TT / QT);
  const int h   = tmp % HH;
  const int b   = tmp / HH;
  const int q0  = qt * QT;
  const int wave = threadIdx.x >> 5;
  const int lane = threadIdx.x & 31;

  const float gamma = gamma_p[0], delta = delta_p[0];
  const int   addb  = addb_p[0];
  const float ap = alpha_pos[h], an = alpha_neg[h];
  const float scale = 0.125f;  // HD^-0.5

  const float* Qb = Qp + (size_t)b * TT * DD + h * HD;
  const float* Kb = Kp + (size_t)b * TT * DD + h * HD;

  // ---- Pass 1: S = scale * Q K^T via WMMA (8 waves = 2 q-halves x 4 k-groups)
  {
    const int qh = wave & 1;
    const int kg = wave >> 1;
    const float* Qrow = Qb + (size_t)(q0 + qh * 16) * DD;
    v16h aq0 = load_a_frag_f32(Qrow +  0, DD);      // hd 0..31
    v16h aq1 = load_a_frag_f32(Qrow + 32, DD);      // hd 32..63
    const int n  = lane & 15;
    const int rb = (lane >> 4) << 3;
    for (int kt = kg; kt < TT / 16; kt += 4) {
      const float* Kt = Kb + (size_t)(kt * 16) * DD;
      v8f c = {};
      c = wmma16(aq0, load_bt_frag_f32(Kt +  0, DD), c);
      c = wmma16(aq1, load_bt_frag_f32(Kt + 32, DD), c);
#pragma unroll
      for (int r = 0; r < 8; ++r)
        S[(qh * 16 + rb + r) * TT + kt * 16 + n] = c[r] * scale;
    }
  }
  __syncthreads();

  // ---- Row stats (mean / 1/(std+eps), ddof=1) for logits and stoich bias
  const int row = threadIdx.x >> 3;   // 0..31
  const int seg = threadIdx.x & 7;    // 0..7 -> 128 cols each
  {
    float s = 0.f, s2 = 0.f;
    const float* Sr = S + row * TT + seg * 128;
    for (int k = 0; k < 128; ++k) { float v = Sr[k]; s += v; s2 += v * v; }
    redA[row * 8 + seg] = s; redB[row * 8 + seg] = s2;
  }
  __syncthreads();
  if (threadIdx.x < QT) {
    const int r = threadIdx.x;
    float s = 0.f, s2 = 0.f;
    for (int i = 0; i < 8; ++i) { s += redA[r * 8 + i]; s2 += redB[r * 8 + i]; }
    float mu  = s * (1.f / TT);
    float var = (s2 - (float)TT * mu * mu) * (1.f / (TT - 1));
    muL[r] = mu;
    isL[r] = 1.f / (sqrtf(fmaxf(var, 0.f)) + EPSF);
    const float* st = stats + ((size_t)b * TT + q0 + r) * 4;
    float ms  = (ap * st[0] + an * st[1]) * (1.f / TT);
    float s2s = ap * ap * st[2] + an * an * st[3];
    float vs  = (s2s - (float)TT * ms * ms) * (1.f / (TT - 1));
    muS[r] = ms;
    isS[r] = 1.f / (sqrtf(fmaxf(vs, 0.f)) + EPSF);
  }
  __syncthreads();

  // ---- Pass 2a: attn = gamma*z(logits) + delta*z(stoich); track row max
  const float fi = frac[(size_t)b * TT + q0 + row];
  const float mu = muL[row], il = isL[row], ms = muS[row], iss = isS[row];
  {
    float mx = -3.0e38f;
    float* Sr = S + row * TT + seg * 128;
    const float* fr = frac + (size_t)b * TT + seg * 128;
    for (int k = 0; k < 128; ++k) {
      float fj  = fr[k];
      float dm  = (fj - fi) * fi * fj;
      float stv = (dm > 0.f ? ap : an) * dm;
      float a   = gamma * (Sr[k] - mu) * il;
      if (addb) a += delta * (stv - ms) * iss;
      Sr[k] = a;
      mx = fmaxf(mx, a);
    }
    redA[row * 8 + seg] = mx;
  }
  __syncthreads();
  if (threadIdx.x < QT) {
    const int r = threadIdx.x;
    float mx = redA[r * 8];
    for (int i = 1; i < 8; ++i) mx = fmaxf(mx, redA[r * 8 + i]);
    rmax[r] = mx;
  }
  __syncthreads();

  // ---- Pass 2b: exp + row denominator (probs left unnormalized in LDS)
  {
    const float mx = rmax[row];
    float s = 0.f;
    float* Sr = S + row * TT + seg * 128;
    for (int k = 0; k < 128; ++k) { float e = __expf(Sr[k] - mx); Sr[k] = e; s += e; }
    redA[row * 8 + seg] = s;
  }
  __syncthreads();
  if (threadIdx.x < QT) {
    const int r = threadIdx.x;
    float s = 0.f;
    for (int i = 0; i < 8; ++i) s += redA[r * 8 + i];
    rden[r] = s;
  }
  __syncthreads();

  // ---- Pass 3: O = (P / den) @ V via WMMA (2 q-halves x 4 hd-tiles);
  //      probs A-frags come from LDS (ds_load_b128), V from packed f16 frags.
  {
    const int qh = wave & 1;
    const int ct = wave >> 1;  // 16-wide hd tile
    v8f acc = {};
    for (int k0 = 0; k0 < TT; k0 += 32) {
      v16h a  = load_a_frag_f32(S + (qh * 16) * TT + k0, TT);
      v16h bv = load_b_packed(Vpk,
            (size_t)((b * TT + k0) >> 5) * (DD / 16) + (h * 4 + ct));
      acc = wmma16(a, bv, acc);
    }
    const int n  = lane & 15;
    const int rb = (lane >> 4) << 3;
#pragma unroll
    for (int r = 0; r < 8; ++r) {
      const int orow = qh * 16 + rb + r;
      Ob[((size_t)b * TT + q0 + orow) * DD + h * HD + ct * 16 + n] =
          acc[r] / rden[orow];
    }
  }
}

extern "C" void kernel_launch(void* const* d_in, const int* in_sizes, int n_in,
                              void* d_out, int out_size, void* d_ws, size_t ws_size,
                              hipStream_t stream) {
  const float* query = (const float*)d_in[0];
  const float* key_  = (const float*)d_in[1];
  const float* value = (const float*)d_in[2];
  const float* frac  = (const float*)d_in[3];
  const float* Wq = (const float*)d_in[4];  const float* bq = (const float*)d_in[5];
  const float* Wk = (const float*)d_in[6];  const float* bk = (const float*)d_in[7];
  const float* Wv = (const float*)d_in[8];  const float* bv = (const float*)d_in[9];
  const float* Wo = (const float*)d_in[10]; const float* bo = (const float*)d_in[11];
  const float* alpha_pos = (const float*)d_in[12];
  const float* alpha_neg = (const float*)d_in[13];
  const float* gamma = (const float*)d_in[14];
  const float* delta = (const float*)d_in[15];
  const int*   addb  = (const int*)d_in[16];

  float* ws = (float*)d_ws;
  const size_t TEN = (size_t)BB * TT * DD;   // 4,194,304 elements
  float* Qp    = ws;
  float* Kp    = Qp + TEN;
  float* Vp    = Kp + TEN;
  float* Ob    = Vp + TEN;
  float* stats = Ob + TEN;                   // BB*TT*4 floats
  _Float16* Vpk = (_Float16*)(stats + (size_t)BB * TT * 4);
  _Float16* Wqp = Vpk + TEN;                 // each W pack: 512*512 halves
  _Float16* Wkp = Wqp + (size_t)DD * DD;
  _Float16* Wvp = Wkp + (size_t)DD * DD;
  _Float16* Wop = Wvp + (size_t)DD * DD;

  const int nrows = BB * TT;                 // 8192

  // Pack the four weight matrices into f16 B-fragment layout (L2-resident,
  // reused by every row tile -> inner loops become pure b128 loads).
  {
    const int thrW = (DD / 32) * (DD / 16) * 32;   // 16384
    pack_b_kernel<<<(thrW + 255) / 256, 256, 0, stream>>>(Wq, Wqp, DD);
    pack_b_kernel<<<(thrW + 255) / 256, 256, 0, stream>>>(Wk, Wkp, DD);
    pack_b_kernel<<<(thrW + 255) / 256, 256, 0, stream>>>(Wv, Wvp, DD);
    pack_b_kernel<<<(thrW + 255) / 256, 256, 0, stream>>>(Wo, Wop, DD);
  }

  const int wavesPerGemm = (nrows / 32) * (DD / 64);   // 2048
  dim3 blkP(256);
  dim3 grdP((wavesPerGemm * 32) / 256);                // 256 blocks

  proj_kernel<<<grdP, blkP, 0, stream>>>(query, Wqp, bq, Qp, nrows);
  proj_kernel<<<grdP, blkP, 0, stream>>>(key_,  Wkp, bk, Kp, nrows);
  proj_kernel<<<grdP, blkP, 0, stream>>>(value, Wvp, bv, Vp, nrows);

  // Pack V projection into f16 fragments (re-read 32x per (b,h) -> halves L2
  // traffic and removes all strided loads from the P@V loop).
  {
    const int thrV = (nrows / 32) * (DD / 16) * 32;    // 262144
    pack_b_kernel<<<(thrV + 255) / 256, 256, 0, stream>>>(Vp, Vpk, nrows);
  }

  stoich_stats_kernel<<<(BB * TT + 255) / 256, 256, 0, stream>>>(frac, stats);

  const size_t smemBytes = (size_t)(QT * TT + 2 * QT * 8 + 6 * QT) * sizeof(float);
  (void)hipFuncSetAttribute(reinterpret_cast<const void*>(attn_kernel),
                            hipFuncAttributeMaxDynamicSharedMemorySize,
                            (int)smemBytes);
  attn_kernel<<<BB * HH * (TT / QT), 256, smemBytes, stream>>>(
      Qp, Kp, Vpk, frac, stats, alpha_pos, alpha_neg, gamma, delta, addb, Ob);

  proj_kernel<<<grdP, blkP, 0, stream>>>(Ob, Wop, bo, (float*)d_out, nrows);
}
